// Pattern_Memory_Block_58420145160853
// MI455X (gfx1250) — matrix operand order repaired and checked by reading, not compiled, not run on hardware
//
#include <hip/hip_runtime.h>
#include <hip/hip_bf16.h>

// ---------- constants for this problem instance ----------
#define BB    64
#define CC    256
#define HH    37
#define HW    1369          // 37*37
#define NTOK  87616         // 64*1369
#define DD    256
#define NUMS  512
#define NRT   5476          // NTOK/16 row tiles
#define THRESH 0.5f
#define RATE   0.999f
#define TOPK   10
#define LDSPAD 8            // bf16 elements of pad per row (16B -> 4-bank skew)
#define LSTRIDE (DD + LDSPAD)   // 264 elements per staged row

typedef __bf16 bf16;
typedef __bf16 v16bf __attribute__((ext_vector_type(16)));
typedef __bf16 v8bf  __attribute__((ext_vector_type(8)));
typedef float  v8f   __attribute__((ext_vector_type(8)));

__device__ __forceinline__ bf16 f2bf(float f) {
    unsigned u = __float_as_uint(f);
    unsigned r = u + 0x7fffu + ((u >> 16) & 1u);   // round-to-nearest-even
    unsigned short h = (unsigned short)(r >> 16);
    return __builtin_bit_cast(bf16, h);
}

// ---- CDNA5 async global -> LDS copy (16B per lane), tracked by ASYNCcnt ----
__device__ __forceinline__ void async_g2l_b128(unsigned lds_byte_off, const void* gaddr) {
    asm volatile("global_load_async_to_lds_b128 %0, %1, off"
                 :: "v"(lds_byte_off), "v"(gaddr) : "memory");
}
__device__ __forceinline__ void wait_async0() {
    asm volatile("s_wait_asynccnt 0x0" ::: "memory");
}

// A-fragment: 16x32 bf16, row-major source with row stride 256 elements.
// lane l (half=l>>4, r=l&15): K chunks [8*half,+8) and [16+8*half,+8) of the 32-K block.
__device__ __forceinline__ v16bf load_afrag(const bf16* __restrict__ A, int row0, int kbase, int lane) {
    int half = lane >> 4, r = lane & 15;
    const bf16* p = A + (size_t)(row0 + r) * DD + kbase + half * 8;
    v8bf lo = *(const v8bf*)p;
    v8bf hi = *(const v8bf*)(p + 16);
    return __builtin_shufflevector(lo, hi, 0,1,2,3,4,5,6,7,8,9,10,11,12,13,14,15);
}

// B-fragment from LDS-staged W (row stride LSTRIDE elements, 16B aligned chunks).
// lane l holds column n=l&15, K chunks [8*half,+8) and [16+8*half,+8) of the 32-K block.
__device__ __forceinline__ v16bf load_bfrag_lds(const bf16* sb, int col, int kbase, int lane) {
    int half = lane >> 4;
    const bf16* p = sb + (size_t)col * LSTRIDE + kbase + half * 8;
    v8bf lo = *(const v8bf*)p;
    v8bf hi = *(const v8bf*)(p + 16);
    return __builtin_shufflevector(lo, hi, 0,1,2,3,4,5,6,7,8,9,10,11,12,13,14,15);
}

// cooperative async stage of ROWS x 256 bf16 row-major W into LDS with LSTRIDE padding
template <int ROWS>
__device__ __forceinline__ void stage_codes(const bf16* __restrict__ W, char* smem, int tid) {
    unsigned base = (unsigned)(size_t)smem;
    const int passes = ROWS * 32 / 1024;   // 32 x 16B chunks per 256-elem row
#pragma unroll
    for (int pass = 0; pass < passes; ++pass) {
        int idx = pass * 1024 + tid;
        int row = idx >> 5, ch = idx & 31;
        unsigned lds = base + (unsigned)((row * LSTRIDE + ch * 8) * 2);
        async_g2l_b128(lds, W + (size_t)row * DD + ch * 8);
    }
    wait_async0();
    __syncthreads();
}

// ---------- elementwise helpers ----------
__global__ void k_convert_bf16(const float* __restrict__ src, bf16* __restrict__ dst, int n) {
    int i = blockIdx.x * 256 + threadIdx.x;
    if (i < n) dst[i] = f2bf(src[i]);
}

__global__ void k_zero_f32(float* __restrict__ p, int n) {
    int i = blockIdx.x * 256 + threadIdx.x;
    if (i < n) p[i] = 0.0f;
}

// z[n,d] = freq[b]*freq_W[d] + freq_b[d] + pe[dp[b],d] + pe[i,d] + pe[j,d]   (stored bf16)
__global__ void k_build_z(const float* __restrict__ freq, const int* __restrict__ dpos,
                          const float* __restrict__ pe, const float* __restrict__ fW,
                          const float* __restrict__ fb, bf16* __restrict__ z) {
    size_t idx = (size_t)blockIdx.x * 256 + threadIdx.x;
    if (idx >= (size_t)NTOK * DD) return;
    int d = (int)(idx & (DD - 1));
    int n = (int)(idx >> 8);
    int b = n / HW;
    int p = n % HW;
    int i = p / HH, j = p % HH;
    int dp = dpos[b];
    float v = freq[b] * fW[d] + fb[d] + pe[dp * DD + d] + pe[i * DD + d] + pe[j * DD + d];
    z[idx] = f2bf(v);
}

// ---------- WMMA GEMM (N x 256) = A * W^T + bias, W staged in LDS ----------
// block = 1024 threads (32 waves); each wave owns a 16-row strip over all 16 col tiles.
template <int GELU>
__global__ __launch_bounds__(1024) void k_gemm256(const bf16* __restrict__ A, const bf16* __restrict__ W,
                                                  const float* __restrict__ bias, void* __restrict__ out,
                                                  int nrt) {
    extern __shared__ char smem[];
    const int tid = threadIdx.x;
    stage_codes<DD>(W, smem, tid);               // 256 rows -> 135KB LDS
    const bf16* sb = (const bf16*)smem;

    int wid = blockIdx.x * 32 + (tid >> 5);
    if (wid >= nrt) return;
    int lane = tid & 31;
    int half = lane >> 4, r = lane & 15;
    int row0 = wid * 16;

    v16bf afr[8];
#pragma unroll
    for (int kt = 0; kt < 8; ++kt) afr[kt] = load_afrag(A, row0, kt * 32, lane);

    for (int ct = 0; ct < 16; ++ct) {
        int col = ct * 16 + r;
        v8f c = {};
#pragma unroll
        for (int kt = 0; kt < 8; ++kt) {
            v16bf bf = load_bfrag_lds(sb, col, kt * 32, lane);
            c = __builtin_amdgcn_wmma_f32_16x16x32_bf16(false, afr[kt], false, bf,
                                                        (short)0, c, false, false);
        }
        float bs = bias[col];
#pragma unroll
        for (int k = 0; k < 8; ++k) {
            int row = row0 + k + half * 8;
            float v = c[k] + bs;
            if (GELU) {
                v = 0.5f * v * (1.0f + erff(v * 0.70710678118654752f));  // exact GELU
                ((bf16*)out)[(size_t)row * DD + col] = f2bf(v);
            } else {
                ((float*)out)[(size_t)row * DD + col] = v;
            }
        }
    }
}

// ---------- WMMA score + fused row-argmax: scores = qn * codes^T, codes staged in LDS ----------
__global__ __launch_bounds__(1024) void k_score_argmax(const bf16* __restrict__ qn,
                                                       const bf16* __restrict__ codes,
                                                       int* __restrict__ out_idx,
                                                       float* __restrict__ out_score, int nrt) {
    extern __shared__ char smem[];
    const int tid = threadIdx.x;
    stage_codes<NUMS>(codes, smem, tid);         // 512 rows -> 270KB LDS (<= 320KB/WGP)
    const bf16* sb = (const bf16*)smem;

    int wid = blockIdx.x * 32 + (tid >> 5);
    if (wid >= nrt) return;
    int lane = tid & 31;
    int half = lane >> 4, r = lane & 15;
    int row0 = wid * 16;

    v16bf afr[8];
#pragma unroll
    for (int kt = 0; kt < 8; ++kt) afr[kt] = load_afrag(qn, row0, kt * 32, lane);

    float maxv[8];
    int   maxi[8];
#pragma unroll
    for (int k = 0; k < 8; ++k) { maxv[k] = -__builtin_inff(); maxi[k] = 0; }

    for (int ct = 0; ct < 32; ++ct) {           // 512 codes / 16
        int col = ct * 16 + r;
        v8f c = {};
#pragma unroll
        for (int kt = 0; kt < 8; ++kt) {
            v16bf bf = load_bfrag_lds(sb, col, kt * 32, lane);
            c = __builtin_amdgcn_wmma_f32_16x16x32_bf16(false, afr[kt], false, bf,
                                                        (short)0, c, false, false);
        }
#pragma unroll
        for (int k = 0; k < 8; ++k) {
            float v = c[k];
            if (v > maxv[k]) { maxv[k] = v; maxi[k] = col; }   // col grows -> first-max kept
        }
    }
    // reduce across the 16 lanes of each half (xor 1,2,4,8 stays within the half)
#pragma unroll
    for (int m = 1; m < 16; m <<= 1) {
#pragma unroll
        for (int k = 0; k < 8; ++k) {
            float ov = __shfl_xor(maxv[k], m, 32);
            int   oi = __shfl_xor(maxi[k], m, 32);
            if (ov > maxv[k] || (ov == maxv[k] && oi < maxi[k])) { maxv[k] = ov; maxi[k] = oi; }
        }
    }
    if (r == 0) {
#pragma unroll
        for (int k = 0; k < 8; ++k) {
            int row = row0 + k + half * 8;
            out_idx[row] = maxi[k];
            if (out_score) out_score[row] = maxv[k];
        }
    }
}

// ---------- L2 row-normalize (f32 in -> bf16 out), one wave per 256-wide row ----------
__global__ __launch_bounds__(256) void k_rownorm(const float* __restrict__ in, bf16* __restrict__ out,
                                                 int rows) {
    int wid = (blockIdx.x * blockDim.x + threadIdx.x) >> 5;
    if (wid >= rows) return;
    int lane = threadIdx.x & 31;
    const float* p = in + (size_t)wid * DD;
    float v[8], ss = 0.0f;
#pragma unroll
    for (int i = 0; i < 8; ++i) { v[i] = p[lane + i * 32]; ss += v[i] * v[i]; }
#pragma unroll
    for (int m = 16; m > 0; m >>= 1) ss += __shfl_xor(ss, m, 32);
    float s = 1.0f / fmaxf(sqrtf(ss), 1e-12f);
    bf16* o = out + (size_t)wid * DD;
#pragma unroll
    for (int i = 0; i < 8; ++i) o[lane + i * 32] = f2bf(v[i] * s);
}

// ---------- per-token pattern cosine, mask, segment sums; one wave per token ----------
__global__ __launch_bounds__(256) void k_patscore(const float* __restrict__ x, const float* __restrict__ q,
                                                  const float* __restrict__ pattern,
                                                  const int* __restrict__ t1i, const float* __restrict__ t1s,
                                                  float* __restrict__ negs, float* __restrict__ counts,
                                                  float* __restrict__ sums) {
    int n = (blockIdx.x * blockDim.x + threadIdx.x) >> 5;
    if (n >= NTOK) return;
    int lane = threadIdx.x & 31;
    int b = n / HW, p = n % HW;
    const float* xc = x + (size_t)b * CC * HW + p;        // element c at + c*HW
    int code = t1i[n];
    const float* pr = pattern + (size_t)code * DD;
    float dxx = 0.f, dvv = 0.f, dxv = 0.f;
#pragma unroll
    for (int i = 0; i < 8; ++i) {
        int c = lane + i * 32;
        float xv = xc[(size_t)c * HW];
        float pv = pr[c];
        dxx += xv * xv; dvv += pv * pv; dxv += xv * pv;
    }
#pragma unroll
    for (int m = 16; m > 0; m >>= 1) {
        dxx += __shfl_xor(dxx, m, 32);
        dvv += __shfl_xor(dvv, m, 32);
        dxv += __shfl_xor(dxv, m, 32);
    }
    float sc = dxv / (fmaxf(sqrtf(dxx), 1e-12f) * fmaxf(sqrtf(dvv), 1e-12f));
    bool mask = sc > THRESH;
    if (lane == 0) {
        negs[n] = mask ? __builtin_inff() : t1s[n];
        if (mask) atomicAdd(&counts[code], 1.0f);
    }
    if (mask) {
        const float* qr = q + (size_t)n * DD;
        float* sr = sums + (size_t)code * DD;
#pragma unroll
        for (int i = 0; i < 8; ++i) { int c = lane + i * 32; atomicAdd(&sr[c], qr[c]); }
    }
}

// ---------- EMA update ----------
__global__ void k_ema(const float* __restrict__ pf, const float* __restrict__ counts,
                      const float* __restrict__ sums, float* __restrict__ params1) {
    int j = blockIdx.x, c = threadIdx.x;
    size_t o = (size_t)j * DD + c;
    float cnt = counts[j];
    float v = pf[o];
    if (cnt > 0.0f) v = v * RATE + (sums[o] / fmaxf(cnt, 1.0f)) * (1.0f - RATE);
    params1[o] = v;
}

// ---------- top-10 largest ages (ties -> lower index), single block of 512 ----------
__global__ __launch_bounds__(512) void k_topk_age(const float* __restrict__ age, int* __restrict__ old_idx) {
    __shared__ float sv[NUMS];
    __shared__ int   si[NUMS];
    int t = threadIdx.x;
    float myv = age[t];
    for (int r = 0; r < TOPK; ++r) {
        sv[t] = myv; si[t] = t;
        __syncthreads();
        for (int s = 256; s > 0; s >>= 1) {
            if (t < s) {
                float ov = sv[t + s]; int oi = si[t + s];
                if (ov > sv[t] || (ov == sv[t] && oi < si[t])) { sv[t] = ov; si[t] = oi; }
            }
            __syncthreads();
        }
        int w = si[0];
        if (t == 0) old_idx[r] = w;
        if (t == w) myv = -__builtin_inff();
        __syncthreads();
    }
}

// ---------- top-10 smallest neg-scores (ties -> lower index), single block of 1024 ----------
__global__ __launch_bounds__(1024) void k_topk_neg(const float* __restrict__ negs, int* __restrict__ neg_idx) {
    __shared__ float sv[1024];
    __shared__ int   si[1024];
    __shared__ int   sel[TOPK];
    int t = threadIdx.x;
    for (int r = 0; r < TOPK; ++r) {
        float bv = __builtin_inff();
        int   bi = 0x7fffffff;
        for (int i = t; i < NTOK; i += 1024) {
            bool skip = false;
            for (int j = 0; j < r; ++j) if (sel[j] == i) skip = true;
            if (!skip) {
                float v = negs[i];
                if (v < bv || (v == bv && i < bi)) { bv = v; bi = i; }
            }
        }
        sv[t] = bv; si[t] = bi;
        __syncthreads();
        for (int s = 512; s > 0; s >>= 1) {
            if (t < s) {
                float ov = sv[t + s]; int oi = si[t + s];
                if (ov < sv[t] || (ov == sv[t] && oi < si[t])) { sv[t] = ov; si[t] = oi; }
            }
            __syncthreads();
        }
        if (t == 0) { sel[r] = si[0]; neg_idx[r] = si[0]; }
        __syncthreads();
    }
}

// ---------- build params2 / pattern2 with row replacement ----------
__global__ void k_build2(const float* __restrict__ params1, const float* __restrict__ pattern,
                         const float* __restrict__ q, const float* __restrict__ x,
                         const int* __restrict__ old_idx, const int* __restrict__ neg_idx,
                         float* __restrict__ params2, float* __restrict__ pattern2) {
    int j = blockIdx.x, c = threadIdx.x;
    int k = -1;
    for (int t = 0; t < TOPK; ++t) if (old_idx[t] == j) { k = t; break; }
    size_t o = (size_t)j * DD + c;
    if (k >= 0) {
        int src = neg_idx[k];
        params2[o] = q[(size_t)src * DD + c];
        int b = src / HW, p = src % HW;
        pattern2[o] = x[((size_t)b * CC + c) * HW + p];
    } else {
        params2[o]  = params1[o];
        pattern2[o] = pattern[o];
    }
}

// ---------- final gather: out[b,c,i,j] = pattern2[label[n], c] ----------
__global__ void k_gather(const float* __restrict__ pattern2, const int* __restrict__ label,
                         float* __restrict__ out) {
    size_t idx = (size_t)blockIdx.x * 256 + threadIdx.x;
    if (idx >= (size_t)NTOK * CC) return;
    int j = (int)(idx % HH);
    size_t t = idx / HH;
    int i = (int)(t % HH); t /= HH;
    int c = (int)(t % CC);
    int b = (int)(t / CC);
    int n = b * HW + i * HH + j;
    out[idx] = pattern2[(size_t)label[n] * DD + c];
}

extern "C" void kernel_launch(void* const* d_in, const int* in_sizes, int n_in,
                              void* d_out, int out_size, void* d_ws, size_t ws_size,
                              hipStream_t stream) {
    const float* x    = (const float*)d_in[0];
    const float* freq = (const float*)d_in[1];
    const int*   dpos = (const int*)  d_in[2];
    const float* pe   = (const float*)d_in[3];
    const float* fW   = (const float*)d_in[4];
    const float* fb   = (const float*)d_in[5];
    const float* fc1W = (const float*)d_in[6];
    const float* fc1b = (const float*)d_in[7];
    const float* fc2W = (const float*)d_in[8];
    const float* fc2b = (const float*)d_in[9];
    const float* pfeat = (const float*)d_in[10];
    const float* patt  = (const float*)d_in[11];
    const float* age   = (const float*)d_in[12];
    float* out = (float*)d_out;

    // ---- workspace carve-out (256B aligned) ----
    size_t off = 0;
    char* ws = (char*)d_ws;
    auto alloc = [&](size_t sz) -> void* {
        void* p = ws + off;
        off += (sz + 255) & ~(size_t)255;
        return p;
    };
    float* q       = (float*)alloc((size_t)NTOK * DD * 4);
    bf16*  z       = (bf16*) alloc((size_t)NTOK * DD * 2);   // reused as qn after GEMM1
    bf16*  h       = (bf16*) alloc((size_t)NTOK * DD * 2);
    bf16*  qn      = z;                                      // alias: z dead after GEMM1
    bf16*  fc1bw   = (bf16*) alloc((size_t)DD * DD * 2);
    bf16*  fc2bw   = (bf16*) alloc((size_t)DD * DD * 2);
    bf16*  mn      = (bf16*) alloc((size_t)NUMS * DD * 2);
    bf16*  mn2     = (bf16*) alloc((size_t)NUMS * DD * 2);
    int*   t1i     = (int*)  alloc((size_t)NTOK * 4);
    float* t1s     = (float*)alloc((size_t)NTOK * 4);
    float* negs    = (float*)alloc((size_t)NTOK * 4);
    int*   label   = (int*)  alloc((size_t)NTOK * 4);
    float* counts  = (float*)alloc((size_t)NUMS * 4);
    float* sums    = (float*)alloc((size_t)NUMS * DD * 4);
    float* params1 = (float*)alloc((size_t)NUMS * DD * 4);
    float* params2 = (float*)alloc((size_t)NUMS * DD * 4);
    float* pat2    = (float*)alloc((size_t)NUMS * DD * 4);
    int*   old_idx = (int*)  alloc(64);
    int*   neg_idx = (int*)  alloc(64);
    (void)ws_size; (void)in_sizes; (void)n_in; (void)out_size;

    const int TB = 256;
    const int GB = (NRT + 31) / 32;                      // 172 blocks of 32 waves
    const size_t shm_mlp   = (size_t)DD   * LSTRIDE * 2; // 135,168 B
    const size_t shm_score = (size_t)NUMS * LSTRIDE * 2; // 270,336 B (<= 320KB WGP LDS)

    // weights -> bf16 ; zero accumulators (every call, for graph replay determinism)
    k_convert_bf16<<<(DD * DD + TB - 1) / TB, TB, 0, stream>>>(fc1W, fc1bw, DD * DD);
    k_convert_bf16<<<(DD * DD + TB - 1) / TB, TB, 0, stream>>>(fc2W, fc2bw, DD * DD);
    k_zero_f32<<<(NUMS + TB - 1) / TB, TB, 0, stream>>>(counts, NUMS);
    k_zero_f32<<<(NUMS * DD + TB - 1) / TB, TB, 0, stream>>>(sums, NUMS * DD);

    // z = fe + pos (bf16)
    k_build_z<<<(int)(((size_t)NTOK * DD + TB - 1) / TB), TB, 0, stream>>>(freq, dpos, pe, fW, fb, z);

    // MLP: h = gelu(z @ fc1^T + b1) ; q = h @ fc2^T + b2   (weights LDS-staged via async copy)
    k_gemm256<1><<<GB, 1024, shm_mlp, stream>>>(z, fc1bw, fc1b, (void*)h, NRT);
    k_gemm256<0><<<GB, 1024, shm_mlp, stream>>>(h, fc2bw, fc2b, (void*)q, NRT);

    // normalize q -> qn (bf16, overwrites z) and codes -> mn (bf16)
    k_rownorm<<<(NTOK + 7) / 8, TB, 0, stream>>>(q, qn, NTOK);
    k_rownorm<<<(NUMS + 7) / 8, TB, 0, stream>>>(pfeat, mn, NUMS);

    // score_par = qn @ mn^T, fused argmax -> top1 idx/score (codebook LDS-staged)
    k_score_argmax<<<GB, 1024, shm_score, stream>>>(qn, mn, t1i, t1s, NRT);

    // per-token pattern cosine + mask + segment sums
    k_patscore<<<(NTOK + 7) / 8, TB, 0, stream>>>(x, q, patt, t1i, t1s, negs, counts, sums);

    // EMA blend
    k_ema<<<NUMS, DD, 0, stream>>>(pfeat, counts, sums, params1);

    // top-k selections
    k_topk_age<<<1, NUMS, 0, stream>>>(age, old_idx);
    k_topk_neg<<<1, 1024, 0, stream>>>(negs, neg_idx);

    // row replacement -> params2 / pattern2
    k_build2<<<NUMS, DD, 0, stream>>>(params1, patt, q, x, old_idx, neg_idx, params2, pat2);

    // final read: label = argmax(qn @ l2norm(params2)^T), out = pattern2[label]
    k_rownorm<<<(NUMS + 7) / 8, TB, 0, stream>>>(params2, mn2, NUMS);
    k_score_argmax<<<GB, 1024, shm_score, stream>>>(qn, mn2, label, (int*)nullptr ? nullptr : (float*)nullptr, NRT);
    k_gather<<<(int)(((size_t)NTOK * CC + TB - 1) / TB), TB, 0, stream>>>(pat2, label, out);
}
